// LRU_5763846111447
// MI455X (gfx1250) — matrix operand order repaired and checked by reading, not compiled, not run on hardware
//
#include <hip/hip_runtime.h>
#include <hip/hip_bf16.h>

// ---------------------------------------------------------------------------
// LRU layer for MI455X (gfx1250, wave32, WMMA).
//   Bu   = input @ (gamma*(B_re+iB_im))^T        -> 2 bf16 WMMA GEMMs, f32 acc
//   s_t  = lam*s_{t-1} + Bu_t  (chunked 3-phase fp32 scan, pre-update states)
//   out  = s_re@C_re^T - s_im@C_im^T + input@D^T -> 3 bf16 WMMA GEMMs, f32 acc
// ---------------------------------------------------------------------------

typedef __attribute__((ext_vector_type(16))) __bf16 v16bf;
typedef __attribute__((ext_vector_type(8)))  float  v8f;

#define BSZ   16
#define SEQ   4096
#define INF   256
#define OUTF  256
#define STF   256
#define MTOT  (BSZ * SEQ)          // 65536 rows
#define NCHUNK 16
#define CHLEN  (SEQ / NCHUNK)      // 256

union Frag { v16bf v; float4 q[2]; };

// A-operand: 16x32 bf16 tile at (m0, k0). ISA layout: lane%16 = M row;
// lanes 0-15 hold K {k0..k0+7, k0+16..k0+23}, lanes 16-31 hold {+8, +24}.
__device__ __forceinline__ v16bf load_a(const __bf16* __restrict__ base, int ld,
                                        int m0, int k0, int lane) {
    int row = m0 + (lane & 15);
    int col = k0 + ((lane >> 4) << 3);
    const __bf16* p = base + (size_t)row * ld + col;
    Frag f;
    f.q[0] = *reinterpret_cast<const float4*>(p);
    f.q[1] = *reinterpret_cast<const float4*>(p + 16);
    return f.v;
}

// B-operand: 32x16 bf16 tile (K x N) for column n0+(lane&15); lanes 0-15 hold
// K {k0..k0+7, k0+16..k0+23}, lanes 16-31 hold {+8, +24}. Source matrix
// is stored row-major [N][K] so each lane does two 16-byte loads, 32 bytes apart.
__device__ __forceinline__ v16bf load_b(const __bf16* __restrict__ base, int ld,
                                        int n0, int k0, int lane) {
    int nn  = n0 + (lane & 15);
    int col = k0 + ((lane >> 4) << 3);
    const __bf16* p = base + (size_t)nn * ld + col;
    Frag f;
    f.q[0] = *reinterpret_cast<const float4*>(p);
    f.q[1] = *reinterpret_cast<const float4*>(p + 16);
    return f.v;
}

// ---------------------------------------------------------------- prep kernels
__global__ void k_cvt_input(const float* __restrict__ in, __bf16* __restrict__ out,
                            int n) {
    int i = blockIdx.x * blockDim.x + threadIdx.x;
    int stride = gridDim.x * blockDim.x;
    for (; i < n; i += stride) out[i] = (__bf16)in[i];
}

__global__ void k_prep(const float* __restrict__ nu_log, const float* __restrict__ th_log,
                       const float* __restrict__ gl,
                       const float* __restrict__ Bre, const float* __restrict__ Bim,
                       const float* __restrict__ Cre, const float* __restrict__ Cim,
                       const float* __restrict__ Dm,
                       __bf16* __restrict__ Wre, __bf16* __restrict__ Wim,
                       __bf16* __restrict__ Crb, __bf16* __restrict__ Cib,
                       __bf16* __restrict__ Db,
                       float* __restrict__ lam_re, float* __restrict__ lam_im) {
    int t = blockIdx.x * blockDim.x + threadIdx.x;     // 0 .. 65535
    int r = t >> 8;
    int c = t & 255;
    float g = __expf(gl[r]);
    Wre[r * INF + c] = (__bf16)(g * Bre[r * INF + c]);
    Wim[r * INF + c] = (__bf16)(g * Bim[r * INF + c]);
    Crb[r * STF + c] = (__bf16)( Cre[r * STF + c]);
    Cib[r * STF + c] = (__bf16)(-Cim[r * STF + c]);    // fold minus into weight
    Db [r * INF + c] = (__bf16)( Dm [r * INF + c]);
    if (t < STF) {
        float la = __expf(-__expf(nu_log[t]));
        float th = __expf(th_log[t]);
        lam_re[t] = la * __cosf(th);
        lam_im[t] = la * __sinf(th);
    }
}

// ------------------------------------------------------------- GEMM 1: Bu
__global__ void k_gemm_bu(const __bf16* __restrict__ in_bf,
                          const __bf16* __restrict__ Wre,
                          const __bf16* __restrict__ Wim,
                          float* __restrict__ Bu_re, float* __restrict__ Bu_im) {
    int wave = blockIdx.x * (blockDim.x >> 5) + (threadIdx.x >> 5);
    int lane = threadIdx.x & 31;
    int m0 = wave << 4;
    if (m0 >= MTOT) return;

    for (int nt = 0; nt < STF / 16; ++nt) {
        int n0 = nt << 4;
        v8f acc_re = {};
        v8f acc_im = {};
#pragma unroll
        for (int kc = 0; kc < INF / 32; ++kc) {
            int k0 = kc << 5;
            __builtin_prefetch(in_bf + (size_t)(m0 + (lane & 15)) * INF +
                               ((k0 + 32) & (INF - 1)), 0, 1);
            v16bf a   = load_a(in_bf, INF, m0, k0, lane);
            v16bf bre = load_b(Wre,  INF, n0, k0, lane);
            v16bf bim = load_b(Wim,  INF, n0, k0, lane);
            acc_re = __builtin_amdgcn_wmma_f32_16x16x32_bf16(
                         false, a, false, bre, (short)0, acc_re, false, false);
            acc_im = __builtin_amdgcn_wmma_f32_16x16x32_bf16(
                         false, a, false, bim, (short)0, acc_im, false, false);
        }
        // C/D layout: vgpr j -> row m0 + j + 8*(lane>>4), col n0 + (lane&15)
        int ncol  = n0 + (lane & 15);
        int rbase = m0 + ((lane >> 4) << 3);
#pragma unroll
        for (int j = 0; j < 8; ++j) {
            Bu_re[(size_t)(rbase + j) * STF + ncol] = acc_re[j];
            Bu_im[(size_t)(rbase + j) * STF + ncol] = acc_im[j];
        }
    }
}

// ----------------------------------------------------- scan phase 1: summaries
__global__ void k_scan_partial(const float* __restrict__ Bu_re,
                               const float* __restrict__ Bu_im,
                               const float* __restrict__ lam_re,
                               const float* __restrict__ lam_im,
                               float* __restrict__ sum_re, float* __restrict__ sum_im) {
    int b = blockIdx.x >> 4;
    int c = blockIdx.x & 15;
    int n = threadIdx.x;
    float lr = lam_re[n], li = lam_im[n];
    float sr = 0.f, si = 0.f;
    size_t base = ((size_t)(b * SEQ + c * CHLEN)) * STF + n;
    for (int j = 0; j < CHLEN; ++j) {
        float br = Bu_re[base + (size_t)j * STF];
        float bi = Bu_im[base + (size_t)j * STF];
        float nr = fmaf(lr, sr, fmaf(-li, si, br));
        float ni = fmaf(lr, si, fmaf( li, sr, bi));
        sr = nr; si = ni;
    }
    sum_re[(b * NCHUNK + c) * STF + n] = sr;
    sum_im[(b * NCHUNK + c) * STF + n] = si;
}

// -------------------------------------------------- scan phase 2: chunk carries
__global__ void k_scan_carry(const float* __restrict__ sum_re,
                             const float* __restrict__ sum_im,
                             const float* __restrict__ lam_re,
                             const float* __restrict__ lam_im,
                             float* __restrict__ car_re, float* __restrict__ car_im) {
    int t = blockIdx.x * blockDim.x + threadIdx.x;     // 0 .. 4095
    int b = t >> 8, n = t & 255;
    float pr = lam_re[n], pi = lam_im[n];
#pragma unroll
    for (int s = 0; s < 8; ++s) {                      // lam^256 by squaring
        float nr = pr * pr - pi * pi;
        float ni = 2.f * pr * pi;
        pr = nr; pi = ni;
    }
    float cr = 0.f, ci = 0.f;
#pragma unroll
    for (int c = 0; c < NCHUNK; ++c) {
        int idx = (b * NCHUNK + c) * STF + n;
        car_re[idx] = cr;                              // exclusive carry
        car_im[idx] = ci;
        float sr = sum_re[idx], si = sum_im[idx];
        float nr = fmaf(pr, cr, fmaf(-pi, ci, sr));
        float ni = fmaf(pr, ci, fmaf( pi, cr, si));
        cr = nr; ci = ni;
    }
}

// ------------------------------- scan phase 3: emit pre-update states (zero @0)
__global__ void k_scan_final(const float* __restrict__ Bu_re,
                             const float* __restrict__ Bu_im,
                             const float* __restrict__ lam_re,
                             const float* __restrict__ lam_im,
                             const float* __restrict__ car_re,
                             const float* __restrict__ car_im,
                             float2* __restrict__ states_out,     // interleaved c64
                             __bf16* __restrict__ sre_bf, __bf16* __restrict__ sim_bf) {
    int b = blockIdx.x >> 4;
    int c = blockIdx.x & 15;
    int n = threadIdx.x;
    float lr = lam_re[n], li = lam_im[n];
    int cidx = (b * NCHUNK + c) * STF + n;
    float sr = car_re[cidx];
    float si = car_im[cidx];
    size_t mbase = (size_t)(b * SEQ + c * CHLEN);
    for (int j = 0; j < CHLEN; ++j) {
        size_t m = mbase + j;
        states_out[m * STF + n] = make_float2(sr, si);
        sre_bf[m * STF + n] = (__bf16)sr;
        sim_bf[m * STF + n] = (__bf16)si;
        float br = Bu_re[m * STF + n];
        float bi = Bu_im[m * STF + n];
        float nr = fmaf(lr, sr, fmaf(-li, si, br));
        float ni = fmaf(lr, si, fmaf( li, sr, bi));
        sr = nr; si = ni;
    }
}

// --------------------------------------------------------- GEMM 2: output proj
__global__ void k_gemm_out(const __bf16* __restrict__ sre_bf,
                           const __bf16* __restrict__ sim_bf,
                           const __bf16* __restrict__ in_bf,
                           const __bf16* __restrict__ Crb,
                           const __bf16* __restrict__ Cib,
                           const __bf16* __restrict__ Db,
                           float* __restrict__ out) {
    int wave = blockIdx.x * (blockDim.x >> 5) + (threadIdx.x >> 5);
    int lane = threadIdx.x & 31;
    int m0 = wave << 4;
    if (m0 >= MTOT) return;

    for (int ot = 0; ot < OUTF / 16; ++ot) {
        int o0 = ot << 4;
        v8f acc = {};
#pragma unroll
        for (int kc = 0; kc < STF / 32; ++kc) {
            int k0 = kc << 5;
            v16bf a1 = load_a(sre_bf, STF, m0, k0, lane);
            v16bf b1 = load_b(Crb,    STF, o0, k0, lane);
            acc = __builtin_amdgcn_wmma_f32_16x16x32_bf16(
                      false, a1, false, b1, (short)0, acc, false, false);
            v16bf a2 = load_a(sim_bf, STF, m0, k0, lane);
            v16bf b2 = load_b(Cib,    STF, o0, k0, lane);
            acc = __builtin_amdgcn_wmma_f32_16x16x32_bf16(
                      false, a2, false, b2, (short)0, acc, false, false);
            v16bf a3 = load_a(in_bf,  INF, m0, k0, lane);
            v16bf b3 = load_b(Db,     INF, o0, k0, lane);
            acc = __builtin_amdgcn_wmma_f32_16x16x32_bf16(
                      false, a3, false, b3, (short)0, acc, false, false);
        }
        int ocol  = o0 + (lane & 15);
        int rbase = m0 + ((lane >> 4) << 3);
#pragma unroll
        for (int j = 0; j < 8; ++j)
            out[(size_t)(rbase + j) * OUTF + ocol] = acc[j];
    }
}

// ---------------------------------------------------------------------- launch
extern "C" void kernel_launch(void* const* d_in, const int* in_sizes, int n_in,
                              void* d_out, int out_size, void* d_ws, size_t ws_size,
                              hipStream_t stream) {
    (void)in_sizes; (void)n_in; (void)out_size; (void)ws_size;

    const float* input   = (const float*)d_in[0];
    const float* nu_log  = (const float*)d_in[1];
    const float* th_log  = (const float*)d_in[2];
    const float* gl      = (const float*)d_in[3];
    const float* Bre     = (const float*)d_in[4];
    const float* Bim     = (const float*)d_in[5];
    const float* Cre     = (const float*)d_in[6];
    const float* Cim     = (const float*)d_in[7];
    const float* Dm      = (const float*)d_in[8];

    float* out_y      = (float*)d_out;                           // [B,L,OUTF] f32
    float2* out_state = (float2*)((char*)d_out +
                        (size_t)MTOT * OUTF * sizeof(float));    // [B,L,STF] c64

    // workspace layout
    char* w = (char*)d_ws;
    size_t off = 0;
    auto take = [&](size_t bytes) { char* p = w + off; off += (bytes + 255) & ~(size_t)255; return p; };
    __bf16* in_bf  = (__bf16*)take((size_t)MTOT * INF * 2);
    __bf16* Wre    = (__bf16*)take((size_t)STF  * INF * 2);
    __bf16* Wim    = (__bf16*)take((size_t)STF  * INF * 2);
    __bf16* Crb    = (__bf16*)take((size_t)OUTF * STF * 2);
    __bf16* Cib    = (__bf16*)take((size_t)OUTF * STF * 2);
    __bf16* Db     = (__bf16*)take((size_t)OUTF * INF * 2);
    float*  lam_re = (float*) take((size_t)STF * 4);
    float*  lam_im = (float*) take((size_t)STF * 4);
    float*  Bu_re  = (float*) take((size_t)MTOT * STF * 4);
    float*  Bu_im  = (float*) take((size_t)MTOT * STF * 4);
    __bf16* sre_bf = (__bf16*)take((size_t)MTOT * STF * 2);
    __bf16* sim_bf = (__bf16*)take((size_t)MTOT * STF * 2);
    float*  sum_re = (float*) take((size_t)BSZ * NCHUNK * STF * 4);
    float*  sum_im = (float*) take((size_t)BSZ * NCHUNK * STF * 4);
    float*  car_re = (float*) take((size_t)BSZ * NCHUNK * STF * 4);
    float*  car_im = (float*) take((size_t)BSZ * NCHUNK * STF * 4);

    const int TPB = 256;

    k_cvt_input<<<4096, TPB, 0, stream>>>(input, in_bf, MTOT * INF);
    k_prep<<<(STF * INF) / TPB, TPB, 0, stream>>>(nu_log, th_log, gl, Bre, Bim,
                                                  Cre, Cim, Dm, Wre, Wim, Crb, Cib,
                                                  Db, lam_re, lam_im);
    // 4096 wave-tiles of 16 rows, 8 waves/block
    k_gemm_bu<<<(MTOT / 16) / 8, TPB, 0, stream>>>(in_bf, Wre, Wim, Bu_re, Bu_im);

    k_scan_partial<<<BSZ * NCHUNK, STF, 0, stream>>>(Bu_re, Bu_im, lam_re, lam_im,
                                                     sum_re, sum_im);
    k_scan_carry<<<(BSZ * STF) / TPB, TPB, 0, stream>>>(sum_re, sum_im, lam_re,
                                                        lam_im, car_re, car_im);
    k_scan_final<<<BSZ * NCHUNK, STF, 0, stream>>>(Bu_re, Bu_im, lam_re, lam_im,
                                                   car_re, car_im, out_state,
                                                   sre_bf, sim_bf);

    k_gemm_out<<<(MTOT / 16) / 8, TPB, 0, stream>>>(sre_bf, sim_bf, in_bf,
                                                    Crb, Cib, Db, out_y);
}
